// QuantizedLinearLayerWithActivation_45329084842316
// MI455X (gfx1250) — matrix operand order repaired and checked by reading, not compiled, not run on hardware
//
#include <hip/hip_runtime.h>

typedef __attribute__((ext_vector_type(16))) _Float16 v16h;
typedef __attribute__((ext_vector_type(8)))  float    v8f;
typedef __attribute__((ext_vector_type(4)))  float    f32x4;
typedef __attribute__((ext_vector_type(4)))  int      i32x4;

#define K_DIM 8192
#define O_DIM 8192
#define M_TOK 16
#define NWAVE 4                      // waves per block, K-split 4 ways
#define K_SLICE (K_DIM / NWAVE)      // 2048 -> 64 WMMA iterations per wave

// Block = 4 waves, all working on the same 16 output channels (obase..obase+15),
// each wave consuming a K-quarter with V_WMMA_F32_16X16X32_F16.
// Partial accumulators + partial x-row-sums are reduced through LDS; wave 0
// applies the (dot - zp*rowsum)*scale + bias epilogue.
__global__ __launch_bounds__(128) void qlinear_wmma_kernel(
    const float* __restrict__ x,      // [16, 8192] fp32
    const int*   __restrict__ w,      // [8192, 8192] int32 (int8-range)
    const float* __restrict__ scale,  // [8192]
    const float* __restrict__ zp,     // [8192]
    const float* __restrict__ bias,   // [8192]
    float*       __restrict__ out)    // [16, 8192]
{
  __shared__ float lacc[NWAVE][32][9];   // padded: (lane*9+j) % 64 conflict-free
  __shared__ float lsum[NWAVE][16];

  const int tid   = threadIdx.x;
  const int wq    = tid >> 5;        // wave id in block: K-slice index
  const int lane  = tid & 31;
  const int half  = lane >> 4;       // 0: lanes 0-15, 1: lanes 16-31
  const int r     = lane & 15;       // A: token row M ; B: output column N
  const int obase = blockIdx.x * 16;

  const int kbeg = wq * K_SLICE;
  const int kend = kbeg + K_SLICE;

  // A-matrix (16-bit, 16x32): lane holds row M=r,
  //   K offsets = {8h+0..8h+7} U {16+8h+0..16+8h+7}
  const float* xp = x + (size_t)r * K_DIM + half * 8;
  // B-matrix (16-bit, 32x16): lane holds column N=r,
  //   K offsets = {8h+0..8h+7} U {16+8h+0..16+8h+7}
  const int* wp = w + (size_t)(obase + r) * K_DIM + half * 8;

  v8f   acc = {};
  float sA  = 0.0f;   // partial row-sum of x (exact fp32, pre-f16-conversion)

#pragma unroll 2
  for (int k0 = kbeg; k0 < kend; k0 += 32) {
    // Prefetch the weight stream ~8 iterations ahead (global_prefetch_b8).
    __builtin_prefetch((const void*)(wp + k0 + 256), 0, 0);

    // ---- load A tile (2x 32B contiguous per lane) ----
    f32x4 a0 = *(const f32x4*)(xp + k0);
    f32x4 a1 = *(const f32x4*)(xp + k0 + 4);
    f32x4 a2 = *(const f32x4*)(xp + k0 + 16);
    f32x4 a3 = *(const f32x4*)(xp + k0 + 20);

    // ---- load B tile (2x 32B contiguous per lane) ----
    i32x4 b0 = *(const i32x4*)(wp + k0);
    i32x4 b1 = *(const i32x4*)(wp + k0 + 4);
    i32x4 b2 = *(const i32x4*)(wp + k0 + 16);
    i32x4 b3 = *(const i32x4*)(wp + k0 + 20);

    v16h A, B;
#pragma unroll
    for (int e = 0; e < 4; ++e) {
      A[e]      = (_Float16)a0[e];
      A[4 + e]  = (_Float16)a1[e];
      A[8 + e]  = (_Float16)a2[e];
      A[12 + e] = (_Float16)a3[e];
      // int8-range: low 16 bits as i16 are exact -> v_cvt_f16_i16 path
      B[e]      = (_Float16)(short)b0[e];
      B[4 + e]  = (_Float16)(short)b1[e];
      B[8 + e]  = (_Float16)(short)b2[e];
      B[12 + e] = (_Float16)(short)b3[e];
      sA += a0[e] + a1[e] + a2[e] + a3[e];
    }

    // D = A x B + C  (fp32 accumulate)
    acc = __builtin_amdgcn_wmma_f32_16x16x32_f16(
        /*neg_a=*/false, A, /*neg_b=*/false, B,
        /*c_mod=*/(short)0, acc, /*reuse_a=*/false, /*reuse_b=*/false);
  }

  // Within-wave row sums: lanes r and r+16 hold complementary K-halves.
  float stot = sA + __shfl_xor(sA, 16, 32);   // lane L: partial rowsum(row L&15)

  // ---- cross-wave reduction through LDS ----
#pragma unroll
  for (int j = 0; j < 8; ++j) lacc[wq][lane][j] = acc[j];
  if (lane < 16) lsum[wq][lane] = stot;
  __syncthreads();

  if (wq == 0) {
    v8f   t = {};
    float S = 0.0f;
#pragma unroll
    for (int q = 0; q < NWAVE; ++q) {
#pragma unroll
      for (int j = 0; j < 8; ++j) t[j] += lacc[q][lane][j];
      S += lsum[q][r];
    }
    // S on lane L = full rowsum of row L&15.

    // Epilogue: out[m,o] = (dot - zp[o]*rowsum(m)) * scale[o] + bias[o]
    const int   o  = obase + r;
    const float sc = scale[o];
    const float z  = zp[o];
    const float bi = bias[o];

#pragma unroll
    for (int j = 0; j < 8; ++j) {
      const int   m  = j + 8 * half;        // C/D layout: lanes 16-31 hold M=8..15
      const float Sm = __shfl(S, m, 32);    // rowsum of token m (from lane m)
      out[(size_t)m * O_DIM + o] = (t[j] - z * Sm) * sc + bi;
    }
  }
}

extern "C" void kernel_launch(void* const* d_in, const int* in_sizes, int n_in,
                              void* d_out, int out_size, void* d_ws, size_t ws_size,
                              hipStream_t stream) {
  const float* x     = (const float*)d_in[0];
  const int*   w     = (const int*)  d_in[1];
  const float* scale = (const float*)d_in[2];
  const float* zp    = (const float*)d_in[3];
  const float* bias  = (const float*)d_in[4];
  float*       out   = (float*)d_out;
  (void)in_sizes; (void)n_in; (void)out_size; (void)d_ws; (void)ws_size;

  // 512 blocks x 4 waves = 2048 waves; one 16-channel tile per block.
  qlinear_wmma_kernel<<<O_DIM / 16, 128, 0, stream>>>(x, w, scale, zp, bias, out);
}
